// DictNet_4002909520042
// MI455X (gfx1250) — matrix operand order repaired and checked by reading, not compiled, not run on hardware
//
#include <hip/hip_runtime.h>
#include <hip/hip_bf16.h>
#include <stdint.h>

typedef __attribute__((ext_vector_type(16))) _Float16 v16h;
typedef __attribute__((ext_vector_type(8)))  _Float16 v8h;
typedef __attribute__((ext_vector_type(8)))  float    v8f;
typedef __attribute__((ext_vector_type(4)))  float    v4f;
typedef __attribute__((ext_vector_type(2)))  float    v2f;

#define NN 3000          // N
#define NP 3008          // N padded to 16 (= 47*64 = 94*32)
#define DD 512           // d
#define FF 14            // filters
#define NC 7             // classes
#define NG 200           // neg groups
#define GG 7             // group size

// -------- workspace layout (bytes) --------
#define OFF_A16   ((size_t)0)                              // f16 [NP x NP]
#define OFF_XT16  (OFF_A16  + (size_t)NP*NP*2)             // f16 [DD x NP]  (x transposed)
#define OFF_YF    (OFF_XT16 + (size_t)DD*NP*2)             // f32 [NP x DD]  (y_hat)
#define OFF_Y16   (OFF_YF   + (size_t)NP*DD*4)             // f16 [NP x DD]
#define OFF_NRM   (OFF_Y16  + (size_t)NP*DD*2)             // f32 [NP]       (row sq-norms)
#define OFF_ACC   (OFF_NRM  + (size_t)NP*4)                // S[7] f32, gL1 f32, cnt[7] i32

// ---------------- init: zero accumulators (graph-replay safe) ----------------
__global__ void k_init(int* acc) {
    if (threadIdx.x < 16) acc[threadIdx.x] = 0;
}

// ---------------- A = I - D@C, streamed NT, emitted as f16 -------------------
__global__ void __launch_bounds__(256) k_buildA(const float* __restrict__ D,
                                                const float* __restrict__ C,
                                                _Float16* __restrict__ A) {
    int m = blockIdx.x * 256 + threadIdx.x;
    int n = blockIdx.y;
    if (m >= NP) return;
    _Float16 out = (_Float16)0.0f;
    if (n < NN && m < NN) {
        float c[FF];
        #pragma unroll
        for (int f = 0; f < FF; ++f) c[f] = C[f];      // uniform -> s_load
        const v2f* p = (const v2f*)(D + ((size_t)n * NN + (size_t)m) * FF); // 8B aligned
        float s = 0.0f;
        #pragma unroll
        for (int k = 0; k < FF / 2; ++k) {
            v2f d = __builtin_nontemporal_load(p + k); // 504MB stream: don't pollute L2
            s += d.x * c[2 * k] + d.y * c[2 * k + 1];
        }
        out = (_Float16)(((n == m) ? 1.0f : 0.0f) - s);
    }
    A[(size_t)n * NP + m] = out;
}

// ---------------- xT (f16, zero-padded rows) ---------------------------------
__global__ void __launch_bounds__(256) k_xT(const float* __restrict__ x,
                                            _Float16* __restrict__ xT) {
    int r = blockIdx.x * 256 + threadIdx.x;
    int c = blockIdx.y;
    if (r >= NP) return;
    float v = (r < NN) ? x[(size_t)r * DD + c] : 0.0f;
    xT[(size_t)c * NP + r] = (_Float16)v;
}

// ---------------- WMMA fragment loaders (16-bit, contiguous-k order) ---------------
// A 16x32: lane holds row m=lane%16; K = b..b+15 contiguous, b = lane<16?0:16
__device__ __forceinline__ v16h fragA(const _Float16* base, int ld, int row0,
                                      int k0, int lane) {
    int r = row0 + (lane & 15);
    int b = (lane & 16) ? 16 : 0;
    const _Float16* p = base + (size_t)r * ld + k0 + b;
    v8h lo = *(const v8h*)p;
    v8h hi = *(const v8h*)(p + 8);
    return __builtin_shufflevector(lo, hi, 0,1,2,3,4,5,6,7,8,9,10,11,12,13,14,15);
}
// B 32x16: lane holds column n=lane%16; K = kk..kk+15 contiguous, kk = lane<16?0:16
__device__ __forceinline__ v16h fragB(const _Float16* base, int ld, int col0,
                                      int k0, int lane) {
    int r  = col0 + (lane & 15);
    int kk = k0 + ((lane & 16) ? 16 : 0);
    return *(const v16h*)(base + (size_t)r * ld + kk);
}

#define WMMA(acc, a, b) \
    acc = __builtin_amdgcn_wmma_f32_16x16x32_f16(false, a, false, b, (short)0, acc, false, false)

// ---------------- GEMM1: Y = A @ x  (wave tile 64x32, 8 WMMA/step) -----------
__global__ void __launch_bounds__(256) k_gemm1(const _Float16* __restrict__ A,
                                               const _Float16* __restrict__ xT,
                                               float* __restrict__ Y) {
    const int lane = threadIdx.x & 31;
    const int wave = threadIdx.x >> 5;
    const int gw = blockIdx.x * 8 + wave;
    const int MW = NP / 64;   // 47
    const int NW = DD / 32;   // 16
    if (gw >= MW * NW) return;
    const int row0 = (gw % MW) * 64;
    const int col0 = (gw / MW) * 32;
    v8f z = {};
    v8f acc[4][2];
    #pragma unroll
    for (int i = 0; i < 4; ++i)
        #pragma unroll
        for (int j = 0; j < 2; ++j) acc[i][j] = z;

    for (int k0 = 0; k0 < NP; k0 += 32) {
        v16h a0 = fragA(A, NP, row0,      k0, lane);
        v16h a1 = fragA(A, NP, row0 + 16, k0, lane);
        v16h a2 = fragA(A, NP, row0 + 32, k0, lane);
        v16h a3 = fragA(A, NP, row0 + 48, k0, lane);
        v16h b0 = fragB(xT, NP, col0,      k0, lane);
        v16h b1 = fragB(xT, NP, col0 + 16, k0, lane);
        WMMA(acc[0][0], a0, b0); WMMA(acc[0][1], a0, b1);
        WMMA(acc[1][0], a1, b0); WMMA(acc[1][1], a1, b1);
        WMMA(acc[2][0], a2, b0); WMMA(acc[2][1], a2, b1);
        WMMA(acc[3][0], a3, b0); WMMA(acc[3][1], a3, b1);
    }
    const int hi8 = (lane & 16) ? 8 : 0;
    const int cc  = lane & 15;
    #pragma unroll
    for (int i = 0; i < 4; ++i)
        #pragma unroll
        for (int j = 0; j < 2; ++j)
            #pragma unroll
            for (int v = 0; v < 8; ++v)
                Y[(size_t)(row0 + 16 * i + v + hi8) * DD + (col0 + 16 * j + cc)] = acc[i][j][v];
}

// ---------------- row sq-norms + f16 copy ------------------------------------
__global__ void __launch_bounds__(256) k_norm16(const float* __restrict__ Y,
                                                _Float16* __restrict__ Y16,
                                                float* __restrict__ nrm) {
    const int lane = threadIdx.x & 31;
    const int wave = threadIdx.x >> 5;
    const int row = blockIdx.x * 8 + wave;
    if (row >= NP) return;
    const float* p = Y + (size_t)row * DD + lane * 16;
    _Float16*    q = Y16 + (size_t)row * DD + lane * 16;
    float s = 0.0f;
    #pragma unroll
    for (int t = 0; t < 2; ++t) {
        v4f u0 = *(const v4f*)(p + 8 * t);
        v4f u1 = *(const v4f*)(p + 8 * t + 4);
        v8h h = {};
        h[0] = (_Float16)u0.x; h[1] = (_Float16)u0.y; h[2] = (_Float16)u0.z; h[3] = (_Float16)u0.w;
        h[4] = (_Float16)u1.x; h[5] = (_Float16)u1.y; h[6] = (_Float16)u1.z; h[7] = (_Float16)u1.w;
        *(v8h*)(q + 8 * t) = h;
        s += u0.x*u0.x + u0.y*u0.y + u0.z*u0.z + u0.w*u0.w;
        s += u1.x*u1.x + u1.y*u1.y + u1.z*u1.z + u1.w*u1.w;
    }
    for (int off = 16; off > 0; off >>= 1) s += __shfl_xor(s, off, 32);
    if (lane == 0) nrm[row] = s;
}

// ------- fused Gram (WMMA) -> pairwise dist -> class-masked accumulation -----
__global__ void __launch_bounds__(256) k_gram_loss2(const _Float16* __restrict__ Y16,
                                                    const float* __restrict__ nrm,
                                                    const int* __restrict__ y,
                                                    const unsigned char* __restrict__ mask,
                                                    float* __restrict__ S) {
    __shared__ float scl[NC];
    if (threadIdx.x < NC) scl[threadIdx.x] = 0.0f;
    __syncthreads();
    const int lane = threadIdx.x & 31;
    const int wave = threadIdx.x >> 5;
    const int gw = blockIdx.x * 8 + wave;
    const int WI = NP / 64;   // 47
    const int WJ = NP / 32;   // 94
    if (gw < WI * WJ) {
        const int row0 = (gw % WI) * 64;
        const int col0 = (gw / WI) * 32;
        v8f z = {};
        v8f acc[4][2];
        #pragma unroll
        for (int i = 0; i < 4; ++i)
            #pragma unroll
            for (int j = 0; j < 2; ++j) acc[i][j] = z;

        for (int k0 = 0; k0 < DD; k0 += 32) {
            v16h a0 = fragA(Y16, DD, row0,      k0, lane);
            v16h a1 = fragA(Y16, DD, row0 + 16, k0, lane);
            v16h a2 = fragA(Y16, DD, row0 + 32, k0, lane);
            v16h a3 = fragA(Y16, DD, row0 + 48, k0, lane);
            v16h b0 = fragB(Y16, DD, col0,      k0, lane);  // Gram: B col n = Y row n
            v16h b1 = fragB(Y16, DD, col0 + 16, k0, lane);
            WMMA(acc[0][0], a0, b0); WMMA(acc[0][1], a0, b1);
            WMMA(acc[1][0], a1, b0); WMMA(acc[1][1], a1, b1);
            WMMA(acc[2][0], a2, b0); WMMA(acc[2][1], a2, b1);
            WMMA(acc[3][0], a3, b0); WMMA(acc[3][1], a3, b1);
        }
        const int hi8 = (lane & 16) ? 8 : 0;
        const int cj  = lane & 15;
        #pragma unroll
        for (int i = 0; i < 4; ++i)
            #pragma unroll
            for (int j = 0; j < 2; ++j)
                #pragma unroll
                for (int v = 0; v < 8; ++v) {
                    int gi = row0 + 16 * i + v + hi8;
                    int gj = col0 + 16 * j + cj;
                    if (gi < NN && gj < NN) {
                        float d2 = nrm[gi] + nrm[gj] - 2.0f * acc[i][j][v];
                        float dist = sqrtf(fmaxf(d2, 0.0f));
                        if (dist > 0.0f) {
                            int yi = y[gi];
                            if (yi == y[gj] && mask[gi] && mask[gj])
                                atomicAdd(&scl[yi], dist);   // ds_add_f32
                        }
                    }
                }
    }
    __syncthreads();
    if (threadIdx.x < NC) atomicAdd(&S[threadIdx.x], scl[threadIdx.x]);
}

// ---------------- class counts -----------------------------------------------
__global__ void k_count(const int* __restrict__ y, const unsigned char* __restrict__ mask,
                        int* __restrict__ cnt) {
    __shared__ int sc[NC];
    if (threadIdx.x < NC) sc[threadIdx.x] = 0;
    __syncthreads();
    for (int i = threadIdx.x; i < NN; i += 256)
        if (mask[i]) atomicAdd(&sc[y[i]], 1);
    __syncthreads();
    if (threadIdx.x < NC) cnt[threadIdx.x] = sc[threadIdx.x];
}

// ---------------- group loss: one wave per negative-sample group -------------
__global__ void k_loss1(const float* __restrict__ Y, const int* __restrict__ grp,
                        float* __restrict__ gL1) {
    __shared__ float rows[GG][DD];            // 14 KB
    const int lane = threadIdx.x;             // 32 threads
    const int g = blockIdx.x;
    int idx[GG];
    #pragma unroll
    for (int r = 0; r < GG; ++r) idx[r] = grp[g * GG + r];
    #pragma unroll
    for (int r = 0; r < GG; ++r) {
        const float* p = Y + (size_t)idx[r] * DD;
        for (int c = lane; c < DD; c += 32) rows[r][c] = p[c];
    }
    __syncthreads();
    float sum = 0.0f;
    for (int a = 0; a < GG; ++a)
        for (int b = a + 1; b < GG; ++b) {
            float pp = 0.0f;
            #pragma unroll
            for (int t = 0; t < 16; ++t) {
                int c = lane + 32 * t;
                float d = rows[a][c] - rows[b][c];
                pp += d * d;
            }
            for (int off = 16; off > 0; off >>= 1) pp += __shfl_xor(pp, off, 32);
            sum += sqrtf(pp);
        }
    // mean over g*g = 49 entries; off-diagonal pairs counted twice, diag = 0
    if (lane == 0) atomicAdd(gL1, (2.0f * sum) / 49.0f);
}

// ---------------- final scalar combine ---------------------------------------
__global__ void k_final(const float* __restrict__ C, const float* __restrict__ S,
                        const int* __restrict__ cnt, const float* __restrict__ gL1,
                        float* __restrict__ out) {
    if (threadIdx.x != 0 || blockIdx.x != 0) return;
    float l1 = 0.0f, l2 = 0.0f;
    for (int f = 0; f < FF; ++f) { float v = C[f]; l1 += fabsf(v); l2 += v * v; }
    l2 = sqrtf(l2);
    float dims = sqrtf((float)FF);
    float spars = (dims - l1 / l2) / (dims - 1.0f);
    float loss2 = 0.0f;
    for (int c = 0; c < NC; ++c) {
        int cn = cnt[c];
        if (cn > 0) { float f = (float)cn; loss2 += S[c] / (f * f); }
    }
    // total = sparsity + loss2 + (-sum_of_group_means) / beta, beta = NG/NC
    out[0] = spars + loss2 - gL1[0] * ((float)NC / (float)NG);
}

// ------------------------------------------------------------------------------
extern "C" void kernel_launch(void* const* d_in, const int* in_sizes, int n_in,
                              void* d_out, int out_size, void* d_ws, size_t ws_size,
                              hipStream_t stream) {
    (void)in_sizes; (void)n_in; (void)out_size; (void)ws_size;
    const float*         D    = (const float*)d_in[0];
    const float*         C    = (const float*)d_in[1];
    const float*         x    = (const float*)d_in[2];
    const int*           y    = (const int*)d_in[3];
    const unsigned char* mask = (const unsigned char*)d_in[4];
    const int*           grp  = (const int*)d_in[5];

    char* ws = (char*)d_ws;
    _Float16* A16  = (_Float16*)(ws + OFF_A16);
    _Float16* XT16 = (_Float16*)(ws + OFF_XT16);
    float*    YF   = (float*)(ws + OFF_YF);
    _Float16* Y16  = (_Float16*)(ws + OFF_Y16);
    float*    NRM  = (float*)(ws + OFF_NRM);
    float*    Sacc = (float*)(ws + OFF_ACC);   // S[0..6]
    float*    gL1  = Sacc + 7;                 // [7]
    int*      cnt  = (int*)(Sacc + 8);         // [8..14]

    k_init<<<1, 32, 0, stream>>>((int*)Sacc);

    dim3 gA((NP + 255) / 256, NP);
    k_buildA<<<gA, 256, 0, stream>>>(D, C, A16);

    dim3 gX((NP + 255) / 256, DD);
    k_xT<<<gX, 256, 0, stream>>>(x, XT16);

    k_gemm1<<<(47 * 16) / 8, 256, 0, stream>>>(A16, XT16, YF);        // 94 blocks
    k_norm16<<<NP / 8, 256, 0, stream>>>(YF, Y16, NRM);               // 376 blocks
    k_gram_loss2<<<(47 * 94 + 7) / 8, 256, 0, stream>>>(Y16, NRM, y, mask, Sacc); // 553
    k_count<<<1, 256, 0, stream>>>(y, mask, cnt);
    k_loss1<<<NG, 32, 0, stream>>>(YF, grp, gL1);
    k_final<<<1, 32, 0, stream>>>(C, Sacc, cnt, gL1, (float*)d_out);
}
